// CausalAttention_1271310319981
// MI455X (gfx1250) — matrix operand order repaired and checked by reading, not compiled, not run on hardware
//
#include <hip/hip_runtime.h>
#include <hip/hip_bf16.h>

typedef __bf16 bf16;
typedef __attribute__((ext_vector_type(16))) bf16 bf16x16;
typedef __attribute__((ext_vector_type(8)))  bf16 bf16x8;
typedef __attribute__((ext_vector_type(8)))  float floatx8;

#define WMMA_BF16(a, b, c) \
  __builtin_amdgcn_wmma_f32_16x16x32_bf16(false, (a), false, (b), (short)0, (c), false, false)

__device__ __forceinline__ bf16x16 join8(bf16x8 lo, bf16x8 hi) {
  bf16x16 r;
#pragma unroll
  for (int i = 0; i < 8; i++) { r[i] = lo[i]; r[i + 8] = hi[i]; }
  return r;
}

// gfx1250 async global->LDS copy, 16B per lane, ASYNCcnt-tracked.
// INST_OFFSET is added to both the global and the LDS per-lane address.
__device__ __forceinline__ void async_copy_b128(unsigned lds_byte_addr, const void* gptr) {
  asm volatile("global_load_async_to_lds_b128 %0, %1, off"
               :: "v"(lds_byte_addr), "v"((unsigned long long)(uintptr_t)gptr)
               : "memory");
}
__device__ __forceinline__ void async_copy_b128_off16(unsigned lds_byte_addr, const void* gptr) {
  asm volatile("global_load_async_to_lds_b128 %0, %1, off offset:16"
               :: "v"(lds_byte_addr), "v"((unsigned long long)(uintptr_t)gptr)
               : "memory");
}
__device__ __forceinline__ void wait_asynccnt0() {
  asm volatile("s_wait_asynccnt 0x0" ::: "memory");
}

// ---------------------------------------------------------------------------
// f32 -> bf16 elementwise convert
// ---------------------------------------------------------------------------
__global__ __launch_bounds__(256) void cvt_bf16_kernel(const float* __restrict__ src,
                                                       bf16* __restrict__ dst,
                                                       unsigned long long n) {
  unsigned long long i = (unsigned long long)blockIdx.x * 256ull + threadIdx.x;
  if (i < n) dst[i] = (bf16)src[i];
}

// ---------------------------------------------------------------------------
// RMS/L2 norm row kernel: xn = x / max(||x||,1e-12) * sqrt(2048) * gamma  (bf16 out)
// ---------------------------------------------------------------------------
__global__ __launch_bounds__(256) void rmsnorm_kernel(const float* __restrict__ x,
                                                      const float* __restrict__ gamma,
                                                      bf16* __restrict__ xn) {
  const int row = blockIdx.x;
  const float* xr = x + (size_t)row * 2048;
  float ss = 0.f;
  for (int c = threadIdx.x; c < 2048; c += 256) { float v = xr[c]; ss += v * v; }
  __shared__ float red[256];
  red[threadIdx.x] = ss;
  __syncthreads();
  for (int s = 128; s > 0; s >>= 1) {
    if (threadIdx.x < s) red[threadIdx.x] += red[threadIdx.x + s];
    __syncthreads();
  }
  const float scale = 45.254833995939045f / fmaxf(sqrtf(red[0]), 1e-12f); // sqrt(2048)/norm
  bf16* o = xn + (size_t)row * 2048;
  for (int c = threadIdx.x; c < 2048; c += 256) o[c] = (bf16)(xr[c] * scale * gamma[c]);
}

// ---------------------------------------------------------------------------
// Tiled WMMA GEMM: C(MxN,f32) = A(MxK,bf16 row-major) @ B(KxN,bf16 row-major)
// 128x128 block tile, K=32 steps, 256 threads = 8 waves, each wave 64x32.
// A tile staged via gfx1250 async global->LDS DMA; B transposed through VGPRs.
// ---------------------------------------------------------------------------
__global__ __launch_bounds__(256) void gemm_bf16_kernel(const bf16* __restrict__ A,
                                                        const bf16* __restrict__ B,
                                                        float* __restrict__ C,
                                                        int M, int N, int K) {
  __shared__ __align__(16) bf16 As[128 * 32];  // [m][k], k-stride 32
  __shared__ __align__(16) bf16 Bs[128 * 32];  // Bt[n][k], k-stride 32
  const int tid = threadIdx.x;
  const int lane = tid & 31;
  const int wave = tid >> 5;
  const int hi16 = lane >> 4;
  const int mlocal = lane & 15;
  const int mblock = blockIdx.y * 128;
  const int nblock = blockIdx.x * 128;
  const int wm = (wave >> 2) * 64;  // 2 wave-rows
  const int wn = (wave & 3) * 32;   // 4 wave-cols

  floatx8 acc[4][2];
  {
    floatx8 z = {0.f, 0.f, 0.f, 0.f, 0.f, 0.f, 0.f, 0.f};
#pragma unroll
    for (int i = 0; i < 4; i++)
#pragma unroll
      for (int j = 0; j < 2; j++) acc[i][j] = z;
  }

  const int ai = tid * 16;                 // 128*32 elems / 256 thr = 16 each
  const int am = ai >> 5, ak = ai & 31;    // ak in {0,16}
  const int bi = tid * 16;
  const int bk = bi >> 7, bn = bi & 127;   // 16 contiguous along n

  const bf16* ag = A + (size_t)(mblock + am) * K + ak;
  const bf16* bg = B + (size_t)bk * N + nblock + bn;
  const unsigned as_lds = (unsigned)(uintptr_t)&As[ai];   // LDS byte address

  for (int k0 = 0; k0 < K; k0 += 32) {
    // A tile: async DMA straight into LDS (no VGPR round trip, ASYNCcnt-tracked)
    async_copy_b128(as_lds, ag);
    async_copy_b128_off16(as_lds, ag);
    // B tile: load 16 contiguous, transpose into Bt[n][k]
    bf16x8 b0 = *(const bf16x8*)(bg);
    bf16x8 b1 = *(const bf16x8*)(bg + 8);
    if (k0 + 32 < K) {                          // near-scope prefetch of next tiles
      __builtin_prefetch(ag + 32, 0, 3);
      __builtin_prefetch(bg + (size_t)32 * N, 0, 3);
    }
#pragma unroll
    for (int i = 0; i < 8; i++) {
      Bs[(bn + i) * 32 + bk] = b0[i];
      Bs[(bn + 8 + i) * 32 + bk] = b1[i];
    }
    wait_asynccnt0();
    __syncthreads();

    bf16x16 bfrag[2];
#pragma unroll
    for (int nn = 0; nn < 2; nn++) {
      const bf16* p = &Bs[(wn + nn * 16 + mlocal) * 32 + (hi16 << 3)];
      bfrag[nn] = join8(*(const bf16x8*)p, *(const bf16x8*)(p + 16));
    }
    bf16x16 afrag[4];
#pragma unroll
    for (int mm = 0; mm < 4; mm++) {
      const bf16* pa = &As[(wm + mm * 16 + mlocal) * 32 + (hi16 << 3)];
      afrag[mm] = join8(*(const bf16x8*)pa, *(const bf16x8*)(pa + 16));
    }
#pragma unroll
    for (int mm = 0; mm < 4; mm++) {
      acc[mm][0] = WMMA_BF16(afrag[mm], bfrag[0], acc[mm][0]);
      acc[mm][1] = WMMA_BF16(afrag[mm], bfrag[1], acc[mm][1]);
    }
    __syncthreads();
    ag += 32;
    bg += (size_t)32 * N;
  }

  const int mg0 = mblock + wm + (hi16 << 3);
  const int ng0 = nblock + wn + mlocal;
#pragma unroll
  for (int mm = 0; mm < 4; mm++)
#pragma unroll
    for (int nn = 0; nn < 2; nn++) {
      float* cp = C + (size_t)(mg0 + mm * 16) * N + ng0 + nn * 16;
#pragma unroll
      for (int r = 0; r < 8; r++) cp[(size_t)r * N] = acc[mm][nn][r];
    }
}

// ---------------------------------------------------------------------------
// RoPE + split: qkv f32 (4096 x 6144) -> cached_kv f32 (d_out), q/k roped bf16, v bf16
// ---------------------------------------------------------------------------
__global__ __launch_bounds__(256) void rope_split_kernel(const float* __restrict__ qkv,
                                                         float* __restrict__ cached_kv,
                                                         bf16* __restrict__ qb,
                                                         bf16* __restrict__ kb,
                                                         bf16* __restrict__ vb) {
  const size_t KVHALF = 2ull * 16 * 2048 * 128;
  size_t idx = (size_t)blockIdx.x * 256 + threadIdx.x;  // 2*16*2048*64 total
  const int i = (int)(idx & 63); idx >>= 6;
  const int nn = (int)(idx & 2047); idx >>= 11;
  const int h = (int)(idx & 15);
  const int b = (int)(idx >> 4);

  const size_t qrow = ((size_t)b * 2048 + nn) * 6144;
  const int col0 = h * 128 + 2 * i;
  const float q0 = qkv[qrow + col0],        q1 = qkv[qrow + col0 + 1];
  const float k0 = qkv[qrow + 2048 + col0], k1 = qkv[qrow + 2048 + col0 + 1];
  const float v0 = qkv[qrow + 4096 + col0], v1 = qkv[qrow + 4096 + col0 + 1];

  const size_t off = (((size_t)b * 16 + h) * 2048 + nn) * 128 + 2 * i;
  cached_kv[off] = k0;           cached_kv[off + 1] = k1;           // k pre-rope
  cached_kv[KVHALF + off] = v0;  cached_kv[KVHALF + off + 1] = v1;  // v

  const float ang = __powf(10000.f, -(2.f * i) / 128.f) * (float)nn;
  float s, c;
  __sincosf(ang, &s, &c);
  qb[off] = (bf16)(q0 * c - q1 * s); qb[off + 1] = (bf16)(q1 * c + q0 * s);
  kb[off] = (bf16)(k0 * c - k1 * s); kb[off + 1] = (bf16)(k1 * c + k0 * s);
  vb[off] = (bf16)v0;                vb[off + 1] = (bf16)v1;
}

// ---------------------------------------------------------------------------
// Flash-style causal attention, one workgroup = 64 query rows of one (b,h),
// 4 waves x 16 rows. Key blocks of 32. All WMMA bf16 / f32-accum.
// ---------------------------------------------------------------------------
__global__ __launch_bounds__(128) void attn_kernel(const bf16* __restrict__ q,
                                                   const bf16* __restrict__ k,
                                                   const bf16* __restrict__ v,
                                                   bf16* __restrict__ o) {
  const int bh = blockIdx.y;
  const int b = bh >> 4, h = bh & 15;
  const size_t base = (size_t)bh * (2048 * 128);
  const bf16* qp = q + base;
  const bf16* kp = k + base;
  const bf16* vp = v + base;
  const int qbase = blockIdx.x * 64;
  const int tid = threadIdx.x, lane = tid & 31, wave = tid >> 5;
  const int mlocal = lane & 15;
  const int hi16 = lane >> 4;

  __shared__ __align__(16) bf16  Vt[128 * 32];     // transposed V block [d][j]
  __shared__ __align__(16) float Sws[4][16 * 32];  // per-wave S staging
  __shared__ __align__(16) bf16  Pws[4][16 * 32];  // per-wave P (A-layout source)
  __shared__ __align__(16) float Aws[4][16];       // per-row alpha
  __shared__ __align__(16) float Lws[4][16];       // per-row l

  // Q fragments in WMMA A-layout, row = qbase + wave*16 + mlocal
  bf16x16 qf[4];
  {
    const bf16* pq = qp + (size_t)(qbase + wave * 16 + mlocal) * 128 + hi16 * 16;
#pragma unroll
    for (int c = 0; c < 4; c++)
      qf[c] = join8(*(const bf16x8*)(pq + c * 32), *(const bf16x8*)(pq + c * 32 + 8));
  }

  floatx8 Ot[8];
  {
    floatx8 z = {0.f, 0.f, 0.f, 0.f, 0.f, 0.f, 0.f, 0.f};
#pragma unroll
    for (int t = 0; t < 8; t++) Ot[t] = z;
  }
  float m_i = -3.0e38f, l_i = 0.f;          // live in lanes 0..15 (lane == row)
  const int qrow_s = qbase + wave * 16 + mlocal;

  const int nkb = (qbase >> 5) + 2;          // key blocks up to block diagonal
  for (int kbk = 0; kbk < nkb; kbk++) {
    const int j0 = kbk * 32;

    // --- cooperative transpose-stage of V block into LDS ---
    {
      const int jl = tid >> 2;
      const int d0 = (tid & 3) * 32;
      const bf16* vr = vp + (size_t)(j0 + jl) * 128 + d0;
      bf16x8 t0 = *(const bf16x8*)(vr);
      bf16x8 t1 = *(const bf16x8*)(vr + 8);
      bf16x8 t2 = *(const bf16x8*)(vr + 16);
      bf16x8 t3 = *(const bf16x8*)(vr + 24);
#pragma unroll
      for (int i = 0; i < 8; i++) {
        Vt[(d0 + i) * 32 + jl]      = t0[i];
        Vt[(d0 + 8 + i) * 32 + jl]  = t1[i];
        Vt[(d0 + 16 + i) * 32 + jl] = t2[i];
        Vt[(d0 + 24 + i) * 32 + jl] = t3[i];
      }
    }
    __syncthreads();

    // --- S = Q K^T : k rows are contiguous -> direct global B-layout loads ---
    floatx8 s0, s1;
    {
      floatx8 z = {0.f, 0.f, 0.f, 0.f, 0.f, 0.f, 0.f, 0.f};
      s0 = z; s1 = z;
    }
#pragma unroll
    for (int c = 0; c < 4; c++) {
      const bf16* p0 = kp + (size_t)(j0 + mlocal) * 128 + c * 32 + hi16 * 16;
      const bf16* p1 = kp + (size_t)(j0 + 16 + mlocal) * 128 + c * 32 + hi16 * 16;
      bf16x16 kf0 = join8(*(const bf16x8*)p0, *(const bf16x8*)(p0 + 8));
      bf16x16 kf1 = join8(*(const bf16x8*)p1, *(const bf16x8*)(p1 + 8));
      s0 = WMMA_BF16(qf[c], kf0, s0);
      s1 = WMMA_BF16(qf[c], kf1, s1);
    }

    // --- stage S (C-layout -> row-major LDS) ---
#pragma unroll
    for (int r = 0; r < 8; r++) {
      const int m = r + hi16 * 8;
      Sws[wave][m * 32 + mlocal] = s0[r];
      Sws[wave][m * 32 + 16 + mlocal] = s1[r];
    }

    // --- online softmax, one lane per row (same-wave LDS ops are in-order) ---
    if (lane < 16) {
      float mloc = -3.0e38f;
      float sv[32];
#pragma unroll
      for (int cc = 0; cc < 32; cc++) {
        float xv = Sws[wave][lane * 32 + cc] * 0.08838834764831845f; // 1/sqrt(128)
        if (j0 + cc > qrow_s) xv = -3.0e38f;                         // causal mask
        sv[cc] = xv;
        mloc = fmaxf(mloc, xv);
      }
      const float mnew = fmaxf(m_i, mloc);
      float sum = 0.f;
#pragma unroll
      for (int cc = 0; cc < 32; cc++) {
        const float pv = __expf(sv[cc] - mnew);
        sum += pv;
        Pws[wave][lane * 32 + cc] = (bf16)pv;
      }
      const float alpha = __expf(m_i - mnew);
      l_i = l_i * alpha + sum;
      m_i = mnew;
      Aws[wave][lane] = alpha;
    }

    // --- rescale accumulators ---
    {
      const float4 a0 = *(const float4*)&Aws[wave][hi16 * 8];
      const float4 a1 = *(const float4*)&Aws[wave][hi16 * 8 + 4];
#pragma unroll
      for (int t = 0; t < 8; t++) {
        Ot[t][0] *= a0.x; Ot[t][1] *= a0.y; Ot[t][2] *= a0.z; Ot[t][3] *= a0.w;
        Ot[t][4] *= a1.x; Ot[t][5] *= a1.y; Ot[t][6] *= a1.z; Ot[t][7] *= a1.w;
      }
    }

    // --- O += P @ V ---
    const bf16* pp = &Pws[wave][mlocal * 32 + hi16 * 8];
    bf16x16 pf = join8(*(const bf16x8*)pp, *(const bf16x8*)(pp + 16));
#pragma unroll
    for (int t = 0; t < 8; t++) {
      const bf16* pv = &Vt[(t * 16 + mlocal) * 32 + hi16 * 8];
      bf16x16 vf = join8(*(const bf16x8*)pv, *(const bf16x8*)(pv + 16));
      Ot[t] = WMMA_BF16(pf, vf, Ot[t]);
    }
    __syncthreads();
  }

  // --- finalize: divide by l, scatter to [b][n][h*128+d] bf16 ---
  if (lane < 16) Lws[wave][lane] = l_i;
  const float4 l0 = *(const float4*)&Lws[wave][hi16 * 8];
  const float4 l1 = *(const float4*)&Lws[wave][hi16 * 8 + 4];
  const float li[8] = {1.f / l0.x, 1.f / l0.y, 1.f / l0.z, 1.f / l0.w,
                       1.f / l1.x, 1.f / l1.y, 1.f / l1.z, 1.f / l1.w};
#pragma unroll
  for (int r = 0; r < 8; r++) {
    const int row = qbase + wave * 16 + r + hi16 * 8;
    bf16* op = o + (size_t)(b * 2048 + row) * 2048 + h * 128 + mlocal;
#pragma unroll
    for (int t = 0; t < 8; t++) op[t * 16] = (bf16)(Ot[t][r] * li[r]);
  }
}

// ---------------------------------------------------------------------------
// Host launcher
// ---------------------------------------------------------------------------
extern "C" void kernel_launch(void* const* d_in, const int* in_sizes, int n_in,
                              void* d_out, int out_size, void* d_ws, size_t ws_size,
                              hipStream_t stream) {
  (void)in_sizes; (void)n_in; (void)out_size; (void)ws_size;
  const float* x     = (const float*)d_in[0];
  const float* gamma = (const float*)d_in[1];
  const float* wqkv  = (const float*)d_in[2];
  const float* wout  = (const float*)d_in[3];

  char* ws = (char*)d_ws;
  constexpr size_t XN_OFF   = 0;                                  // 4096*2048 bf16
  constexpr size_t WQKV_OFF = XN_OFF   + 4096ull * 2048 * 2;      // 2048*6144 bf16
  constexpr size_t WOUT_OFF = WQKV_OFF + 2048ull * 6144 * 2;      // 2048*2048 bf16
  constexpr size_t QKV_OFF  = WOUT_OFF + 2048ull * 2048 * 2;      // 4096*6144 f32
  constexpr size_t QB_OFF   = QKV_OFF  + 4096ull * 6144 * 4;      // 2*16*2048*128 bf16
  constexpr size_t KB_OFF   = QB_OFF   + 8388608ull * 2;
  constexpr size_t VB_OFF   = KB_OFF   + 8388608ull * 2;
  constexpr size_t OB_OFF   = VB_OFF   + 8388608ull * 2;          // 4096*2048 bf16

  bf16*  xnb   = (bf16*)(ws + XN_OFF);
  bf16*  wqkvb = (bf16*)(ws + WQKV_OFF);
  bf16*  woutb = (bf16*)(ws + WOUT_OFF);
  float* qkv   = (float*)(ws + QKV_OFF);
  bf16*  qb    = (bf16*)(ws + QB_OFF);
  bf16*  kb    = (bf16*)(ws + KB_OFF);
  bf16*  vb    = (bf16*)(ws + VB_OFF);
  bf16*  ob    = (bf16*)(ws + OB_OFF);

  float* out       = (float*)d_out;                 // (2,2048,2048)
  float* cached_kv = out + 2ull * 2048 * 2048;      // (2,2,16,2048,128)

  // 1) weights -> bf16
  cvt_bf16_kernel<<<(2048ull * 6144) / 256, 256, 0, stream>>>(wqkv, wqkvb, 2048ull * 6144);
  cvt_bf16_kernel<<<(2048ull * 2048) / 256, 256, 0, stream>>>(wout, woutb, 2048ull * 2048);
  // 2) L2-norm scaling
  rmsnorm_kernel<<<4096, 256, 0, stream>>>(x, gamma, xnb);
  // 3) QKV projection: (4096x2048)@(2048x6144)
  gemm_bf16_kernel<<<dim3(48, 32), 256, 0, stream>>>(xnb, wqkvb, qkv, 4096, 6144, 2048);
  // 4) RoPE + split + cached_kv
  rope_split_kernel<<<16384, 256, 0, stream>>>(qkv, cached_kv, qb, kb, vb);
  // 5) causal attention
  attn_kernel<<<dim3(32, 32), 128, 0, stream>>>(qb, kb, vb, ob);
  // 6) output projection: (4096x2048)@(2048x2048)
  gemm_bf16_kernel<<<dim3(16, 32), 256, 0, stream>>>(ob, woutb, out, 4096, 2048, 2048);
}
